// EnhancedSeparateDomainModel_62534723830406
// MI455X (gfx1250) — compile-verified, matrix operand order checked by reading, not run on hardware
//
#include <hip/hip_runtime.h>
#include <hip/hip_bf16.h>

typedef __attribute__((ext_vector_type(16))) _Float16 v16h;
typedef __attribute__((ext_vector_type(8)))  _Float16 v8h;
typedef __attribute__((ext_vector_type(8)))  float    v8f;

#define BATCH 32

// ---------------------------------------------------------------------------
// conv1: 1->32 channels, 3x3 SAME, relu.  Direct (K=9, not WMMA-shaped).
// Input NCHW f32 (C=1), output NHWC f16.  4 threads/pixel, 8 channels each,
// so every thread does one coalesced 16B v8h store.
// ---------------------------------------------------------------------------
__global__ __launch_bounds__(256) void conv1_direct_kernel(
    const float* __restrict__ in, const float* __restrict__ w,
    const float* __restrict__ bias, _Float16* __restrict__ out,
    int H, int W)
{
  __shared__ float sw[32 * 9];
  __shared__ float sb[32];
  int tid = threadIdx.x;
  for (int i = tid; i < 32 * 9; i += blockDim.x) sw[i] = w[i];
  if (tid < 32) sb[tid] = bias[tid];
  __syncthreads();

  long long gid = (long long)blockIdx.x * blockDim.x + tid;
  long long total = (long long)BATCH * H * W * 4;  // 4 channel-groups/pixel
  if (gid >= total) return;
  int cg = (int)(gid & 3);
  long long idx = gid >> 2;
  int x = (int)(idx % W);
  int y = (int)((idx / W) % H);
  int b = (int)(idx / ((long long)W * H));

  float px[9];
#pragma unroll
  for (int t = 0; t < 9; ++t) {
    int dy = t / 3 - 1, dx = t % 3 - 1;
    int yy = y + dy, xx = x + dx;
    px[t] = (yy >= 0 && yy < H && xx >= 0 && xx < W)
                ? in[((long long)b * H + yy) * W + xx]
                : 0.f;
  }
  v8h o;
#pragma unroll
  for (int j = 0; j < 8; ++j) {
    int co = cg * 8 + j;
    float a = sb[co];
#pragma unroll
    for (int t = 0; t < 9; ++t) a += px[t] * sw[co * 9 + t];
    a = a > 0.f ? a : 0.f;
    o[j] = (_Float16)a;
  }
  *(v8h*)(out + idx * 32 + cg * 8) = o;
}

// ---------------------------------------------------------------------------
// Prepack OIHW f32 weights -> wB[co][k] f16 with k = tap*Cin + ci.
// ---------------------------------------------------------------------------
__global__ void prepack_w_kernel(const float* __restrict__ w,
                                 _Float16* __restrict__ wB, int Cin, int Cout)
{
  int K = 9 * Cin;
  int idx = blockIdx.x * blockDim.x + threadIdx.x;
  if (idx >= Cout * K) return;
  int co = idx / K, k = idx % K;
  int tap = k / Cin, ci = k % Cin;
  wB[idx] = (_Float16)w[((long long)co * Cin + ci) * 9 + tap];
}

// ---------------------------------------------------------------------------
// Implicit-GEMM 3x3 SAME conv via V_WMMA_F32_16X16X32_F16, fused bias+relu.
// Activations NHWC f16 (square S x S image), weights wB[co][tap*Cin+ci] f16.
// One wave => 16 output pixels x 16 output channels; K = 9*Cin in chunks
// of 32.  ISA 7.12.2 16-bit A striping (lane hi, element j -> K = hi*8+j /
// 16+hi*8+j) + (tap,ci) K-order => all fragments are contiguous 16B loads.
// Wave-uniform interior fast path: A addresses = one base + compile-time
// immediate offsets (no per-chunk address math, no masking).
// ---------------------------------------------------------------------------
template <int CIN, int S>
__global__ __launch_bounds__(128) void conv3x3_wmma_kernel(
    const _Float16* __restrict__ in, const _Float16* __restrict__ wB,
    const float* __restrict__ bias, _Float16* __restrict__ out, int Cout)
{
  constexpr int K = 9 * CIN;
  constexpr int CPT = CIN / 32;  // K-chunks per filter tap

  int ntc = Cout >> 4;
  int xt = (blockIdx.x / ntc) * 4 + threadIdx.y;
  int ct = blockIdx.x % ntc;
  int y = blockIdx.y;
  int b = blockIdx.z;
  int lane = threadIdx.x;
  int m = lane & 15;   // A row (pixel); also N column for B/C/D
  int hi = lane >> 4;  // half-wave selector in A/B striping
  int x0 = xt << 4, co0 = ct << 4;

  v8f acc;
  {
    float bv = bias[co0 + m];
#pragma unroll
    for (int r = 0; r < 8; ++r) acc[r] = bv;
  }

  const _Float16* wrow = wB + (long long)(co0 + m) * K;

  bool interior = (y >= 1) & (y <= S - 2) & (x0 >= 1) & (x0 <= S - 17);
  if (interior) {
    // ---- fast path: no bounds checks, immediate-offset loads ----
    const _Float16* base =
        in + (((long long)b * S + y) * S + x0 + m) * CIN + hi * 8;
#pragma unroll
    for (int chunk = 0; chunk < K / 32; ++chunk) {
      constexpr int dummy = 0; (void)dummy;
      const int kb = chunk * 32;
      const int tap = chunk / CPT;
      const int ci0 = (chunk - tap * CPT) * 32;
      const int dy = tap / 3 - 1;
      const int dx = tap % 3 - 1;
      const int aoff = (dy * S + dx) * CIN + ci0;  // compile-time constant

      v8h a_lo = *(const v8h*)(base + aoff);
      v8h a_hi = *(const v8h*)(base + aoff + 16);
      v8h b_lo = *(const v8h*)(wrow + kb + hi * 8);
      v8h b_hi = *(const v8h*)(wrow + kb + 16 + hi * 8);

      v16h Am, Bm;
#pragma unroll
      for (int j = 0; j < 8; ++j) {
        Am[j] = a_lo[j]; Am[j + 8] = a_hi[j];
        Bm[j] = b_lo[j]; Bm[j + 8] = b_hi[j];
      }
      acc = __builtin_amdgcn_wmma_f32_16x16x32_f16(
          false, Am, false, Bm, (short)0, acc, false, false);
    }
  } else {
    // ---- edge path: per-lane zero padding ----
#pragma unroll
    for (int chunk = 0; chunk < K / 32; ++chunk) {
      const int kb = chunk * 32;
      const int tap = chunk / CPT;
      const int ci0 = (chunk - tap * CPT) * 32;
      const int dy = tap / 3 - 1;
      const int dx = tap % 3 - 1;
      int yy = y + dy;
      int xx = x0 + m + dx;

      v8h a_lo, a_hi;
      if (yy >= 0 && yy < S && xx >= 0 && xx < S) {
        const _Float16* p =
            in + (((long long)b * S + yy) * S + xx) * CIN + ci0 + hi * 8;
        a_lo = *(const v8h*)(p);
        a_hi = *(const v8h*)(p + 16);
      } else {
#pragma unroll
        for (int j = 0; j < 8; ++j) {
          a_lo[j] = (_Float16)0;
          a_hi[j] = (_Float16)0;
        }
      }
      v8h b_lo = *(const v8h*)(wrow + kb + hi * 8);
      v8h b_hi = *(const v8h*)(wrow + kb + 16 + hi * 8);

      v16h Am, Bm;
#pragma unroll
      for (int j = 0; j < 8; ++j) {
        Am[j] = a_lo[j]; Am[j + 8] = a_hi[j];
        Bm[j] = b_lo[j]; Bm[j + 8] = b_hi[j];
      }
      acc = __builtin_amdgcn_wmma_f32_16x16x32_f16(
          false, Am, false, Bm, (short)0, acc, false, false);
    }
  }

  // D layout: lane holds N = lane%16; VGPR r holds pixel M = r + 8*(lane/16).
#pragma unroll
  for (int r = 0; r < 8; ++r) {
    float v = acc[r];
    v = v > 0.f ? v : 0.f;
    int xm = x0 + r + hi * 8;
    out[(((long long)b * S + y) * S + xm) * Cout + co0 + m] = (_Float16)v;
  }
}

// ---------------------------------------------------------------------------
// 2x2 maxpool on NHWC f16.
// ---------------------------------------------------------------------------
__global__ void maxpool2_kernel(const _Float16* __restrict__ in,
                                _Float16* __restrict__ out, int Ho, int Wo,
                                int C)
{
  long long idx = (long long)blockIdx.x * blockDim.x + threadIdx.x;
  long long total = (long long)BATCH * Ho * Wo * C;
  if (idx >= total) return;
  int c = (int)(idx % C);
  long long r = idx / C;
  int x = (int)(r % Wo);
  r /= Wo;
  int y = (int)(r % Ho);
  int b = (int)(r / Ho);
  int Wi = Wo * 2;
  const _Float16* p =
      in + (((long long)b * (Ho * 2) + y * 2) * Wi + x * 2) * C + c;
  float v0 = (float)p[0];
  float v1 = (float)p[C];
  float v2 = (float)p[(long long)Wi * C];
  float v3 = (float)p[(long long)Wi * C + C];
  float mx = fmaxf(fmaxf(v0, v1), fmaxf(v2, v3));
  out[idx] = (_Float16)mx;
}

// ---------------------------------------------------------------------------
// 16x16 block average of (B,64,64,64) NHWC f16 -> feats1024[b][c*16+gy*4+gx].
// Channel index is the fastest thread dimension -> coalesced reads.
// ---------------------------------------------------------------------------
__global__ void avg16_kernel(const _Float16* __restrict__ in,
                             float* __restrict__ out)
{
  int idx = blockIdx.x * blockDim.x + threadIdx.x;
  if (idx >= BATCH * 64 * 16) return;
  int c = idx & 63, gx = (idx >> 6) & 3, gy = (idx >> 8) & 3, b = idx >> 10;
  float s = 0.f;
  for (int yy = 0; yy < 16; ++yy)
    for (int xx = 0; xx < 16; ++xx)
      s += (float)in[(((long long)b * 64 + (gy * 16 + yy)) * 64 +
                      (gx * 16 + xx)) * 64 + c];
  out[b * 1024 + c * 16 + gy * 4 + gx] = s * (1.f / 256.f);
}

// ---------------------------------------------------------------------------
// mean over the 20 patches (all at row 0, cols 8p): mp[b][r*16+c].
// Patch embedding is linear, so mean-then-GEMV == GEMV-then-mean.
// ---------------------------------------------------------------------------
__global__ void mean_patches_kernel(const float* __restrict__ sig,
                                    float* __restrict__ mp)
{
  int idx = blockIdx.x * blockDim.x + threadIdx.x;
  if (idx >= BATCH * 256) return;
  int col = idx & 255, b = idx >> 8;
  int r = col >> 4, c = col & 15;
  float s = 0.f;
  for (int p = 0; p < 20; ++p)
    s += sig[(long long)b * 65536 + r * 256 + p * 8 + c];
  mp[idx] = s * 0.05f;  // /20
}

// ---------------------------------------------------------------------------
// Generic tiny GEMV: out[b][o] = act( x[b]·W[o] + bias[o] ).  All tail layers.
// ---------------------------------------------------------------------------
__global__ void gemv_kernel(const float* __restrict__ x, int ldx,
                            const float* __restrict__ W,
                            const float* __restrict__ bias,
                            float* __restrict__ out, int ldo, int In, int Out,
                            int relu)
{
  int idx = blockIdx.x * blockDim.x + threadIdx.x;
  if (idx >= BATCH * Out) return;
  int o = idx % Out, b = idx / Out;
  const float* xr = x + (long long)b * ldx;
  const float* wr = W + (long long)o * In;
  float s = bias ? bias[o] : 0.f;
  for (int i = 0; i < In; ++i) s += xr[i] * wr[i];
  if (relu && s < 0.f) s = 0.f;
  out[(long long)b * ldo + o] = s;
}

static inline int cdiv_ll(long long a, int b) { return (int)((a + b - 1) / b); }

extern "C" void kernel_launch(void* const* d_in, const int* in_sizes, int n_in,
                              void* d_out, int out_size, void* d_ws,
                              size_t ws_size, hipStream_t stream)
{
  (void)in_sizes; (void)n_in; (void)out_size; (void)ws_size;

  char* ws = (char*)d_ws;
  size_t off = 0;
  auto take = [&](size_t bytes) -> char* {
    char* p = ws + off;
    off += (bytes + 255) & ~(size_t)255;
    return p;
  };

  // Ping-pong NHWC f16 activation buffers (reused across both branches).
  _Float16* act0 = (_Float16*)take((size_t)BATCH * 256 * 256 * 32 * 2);
  _Float16* act1 = (_Float16*)take((size_t)BATCH * 256 * 256 * 32 * 2);
  _Float16* wBbuf = (_Float16*)take((size_t)64 * 576 * 2);
  float* feats1024 = (float*)take((size_t)BATCH * 1024 * 4);
  float* tf = (float*)take((size_t)BATCH * 128 * 4);
  float* tr = (float*)take((size_t)BATCH * 64 * 4);
  float* vb = (float*)take((size_t)BATCH * 64 * 4);
  float* mp = (float*)take((size_t)BATCH * 256 * 4);
  float* feats = (float*)take((size_t)BATCH * 256 * 4);
  float* h1 = (float*)take((size_t)BATCH * 256 * 4);
  float* h2 = (float*)take((size_t)BATCH * 128 * 4);

  auto F = [&](int i) { return (const float*)d_in[i]; };

  auto run_branch = [&](const float* x, const float* c1w, const float* c1b,
                        const float* c2w, const float* c2b, const float* c3w,
                        const float* c3b, const float* c4w, const float* c4b,
                        const float* fcw, const float* fcb, const float* redw,
                        const float* redb, const float* wv, const float* wow,
                        const float* wob, const float* pew, const float* peb,
                        float* ta_dst, float* tp_dst) {
    // conv1 -> act0 (256x256x32 NHWC f16)
    conv1_direct_kernel<<<cdiv_ll((long long)BATCH * 256 * 256 * 4, 256), 256,
                          0, stream>>>(x, c1w, c1b, act0, 256, 256);
    // conv2 (32->32 @256^2), WMMA
    prepack_w_kernel<<<cdiv_ll(32 * 288, 256), 256, 0, stream>>>(c2w, wBbuf,
                                                                 32, 32);
    conv3x3_wmma_kernel<32, 256>
        <<<dim3((256 / 16 / 4) * (32 / 16), 256, BATCH), dim3(32, 4), 0,
           stream>>>(act0, wBbuf, c2b, act1, 32);
    // pool -> act0 (128x128x32)
    maxpool2_kernel<<<cdiv_ll((long long)BATCH * 128 * 128 * 32, 256), 256, 0,
                      stream>>>(act1, act0, 128, 128, 32);
    // conv3 (32->64 @128^2), WMMA
    prepack_w_kernel<<<cdiv_ll(64 * 288, 256), 256, 0, stream>>>(c3w, wBbuf,
                                                                 32, 64);
    conv3x3_wmma_kernel<32, 128>
        <<<dim3((128 / 16 / 4) * (64 / 16), 128, BATCH), dim3(32, 4), 0,
           stream>>>(act0, wBbuf, c3b, act1, 64);
    // conv4 (64->64 @128^2), WMMA
    prepack_w_kernel<<<cdiv_ll(64 * 576, 256), 256, 0, stream>>>(c4w, wBbuf,
                                                                 64, 64);
    conv3x3_wmma_kernel<64, 128>
        <<<dim3((128 / 16 / 4) * (64 / 16), 128, BATCH), dim3(32, 4), 0,
           stream>>>(act1, wBbuf, c4b, act0, 64);
    // pool -> act1 (64x64x64)
    maxpool2_kernel<<<cdiv_ll((long long)BATCH * 64 * 64 * 64, 256), 256, 0,
                      stream>>>(act0, act1, 64, 64, 64);
    // 16x16 avg -> (B,1024) f32
    avg16_kernel<<<cdiv_ll(BATCH * 64 * 16, 256), 256, 0, stream>>>(act1,
                                                                    feats1024);
    // fc: (B,1024)->(B,128), no relu
    gemv_kernel<<<cdiv_ll(BATCH * 128, 256), 256, 0, stream>>>(
        feats1024, 1024, fcw, fcb, tf, 128, 1024, 128, 0);
    // red: relu((B,128)->(B,64))
    gemv_kernel<<<cdiv_ll(BATCH * 64, 256), 256, 0, stream>>>(
        tf, 128, redw, redb, tr, 64, 128, 64, 1);
    // attn collapses (S=1: RoPE identity, softmax==1): ta = (tr@wv.T)@wow.T+wob
    gemv_kernel<<<cdiv_ll(BATCH * 64, 256), 256, 0, stream>>>(
        tr, 64, wv, nullptr, vb, 64, 64, 64, 0);
    gemv_kernel<<<cdiv_ll(BATCH * 64, 256), 256, 0, stream>>>(
        vb, 64, wow, wob, ta_dst, 256, 64, 64, 0);
    // patch features: tp = mean(patches) @ pew.T + peb
    mean_patches_kernel<<<cdiv_ll(BATCH * 256, 256), 256, 0, stream>>>(x, mp);
    gemv_kernel<<<cdiv_ll(BATCH * 64, 256), 256, 0, stream>>>(
        mp, 256, pew, peb, tp_dst, 256, 256, 64, 0);
  };

  // Input index map (setup_inputs order):
  // 0 time, 1 freq; t_*: 2..20; f_*: 21..39; head: 40..45.
  run_branch((const float*)d_in[0], F(2), F(3), F(4), F(5), F(6), F(7), F(8),
             F(9), F(10), F(11), F(12), F(13), F(16), F(17), F(18), F(19),
             F(20), feats + 0, feats + 128);
  run_branch((const float*)d_in[1], F(21), F(22), F(23), F(24), F(25), F(26),
             F(27), F(28), F(29), F(30), F(31), F(32), F(35), F(36), F(37),
             F(38), F(39), feats + 64, feats + 192);

  // classifier head: 256 -> 256 -> 128 -> 2
  gemv_kernel<<<cdiv_ll(BATCH * 256, 256), 256, 0, stream>>>(
      feats, 256, F(40), F(41), h1, 256, 256, 256, 1);
  gemv_kernel<<<cdiv_ll(BATCH * 128, 256), 256, 0, stream>>>(
      h1, 256, F(42), F(43), h2, 128, 256, 128, 1);
  gemv_kernel<<<1, 256, 0, stream>>>(h2, 128, F(44), F(45), (float*)d_out, 2,
                                     128, 2, 0);
}
